// differN_noise_21285857919119
// MI455X (gfx1250) — matrix operand order repaired and checked by reading, not compiled, not run on hardware
//
#include <hip/hip_runtime.h>

#define B_ROWS 8192
#define DIM    1024
#define NWIRE  10

typedef _Float16 v16h __attribute__((ext_vector_type(16)));
typedef _Float16 v8h  __attribute__((ext_vector_type(8)));
typedef float    v8f  __attribute__((ext_vector_type(8)));

__device__ __forceinline__ v8f wmmaF32(v16h a, v16h b, v8f c) {
  return __builtin_amdgcn_wmma_f32_16x16x32_f16(false, a, false, b, (short)0, c, false, false);
}

__device__ __forceinline__ v16h frag_cat(const _Float16* lo, const _Float16* hi) {
  v8h a = *(const v8h*)lo;
  v8h b = *(const v8h*)hi;
  return __builtin_shufflevector(a, b, 0,1,2,3,4,5,6,7,8,9,10,11,12,13,14,15);
}

// ---- CDNA5 async global->LDS copy (16B per lane), tracked by ASYNCcnt ----
__device__ __forceinline__ void async_cp16B(const _Float16* g, _Float16* l) {
  asm volatile("global_load_async_to_lds_b128 %0, %1, off"
               :: "v"((__attribute__((address_space(3))) void*)l),
                  "v"((const __attribute__((address_space(1))) void*)g)
               : "memory");
}

__device__ __forceinline__ void wait_async0() {
  asm volatile("s_wait_asynccnt 0x0" ::: "memory");
}

// ---------------- PCA: column means (two phase) ----------------
__global__ void k_colmeans1(const float* __restrict__ X, float* __restrict__ meansP) {
  int bid = blockIdx.x;                 // 32 blocks
  int c  = (bid & 3) * 256 + threadIdx.x;
  int r0 = (bid >> 2) * 1024;
  float s = 0.f;
  for (int r = 0; r < 1024; ++r) s += X[(size_t)(r0 + r) * DIM + c];
  meansP[(size_t)(bid >> 2) * DIM + c] = s;
}
__global__ void k_colmeans2(const float* __restrict__ meansP, float* __restrict__ means) {
  int c = blockIdx.x * 256 + threadIdx.x;   // 4 blocks
  float s = 0.f;
  for (int g = 0; g < 8; ++g) s += meansP[(size_t)g * DIM + c];
  means[c] = s * (1.f / 8192.f);
}

// ---------------- center + hi/lo f16 split ----------------
__global__ void k_center_split(const float* __restrict__ X, const float* __restrict__ means,
                               _Float16* __restrict__ Xh, _Float16* __restrict__ Xl) {
  size_t idx = (size_t)blockIdx.x * 256 + threadIdx.x;
  int c = (int)(idx & (DIM - 1));
  float v = X[idx] - means[c];
  _Float16 h = (_Float16)v;
  Xh[idx] = h;
  Xl[idx] = (_Float16)(v - (float)h);
}

// ---------------- Gram = Xc^T Xc via split-f16 WMMA ----------------
__global__ void k_gram(const _Float16* __restrict__ Xh, const _Float16* __restrict__ Xl,
                       float* __restrict__ C) {
  __shared__ _Float16 sA[2][128][48];
  __shared__ _Float16 sB[2][32][48];
  const int t = threadIdx.x;
  const int m0 = blockIdx.x * 128, n0 = blockIdx.y * 32;
  const int wv = t >> 5, lane = t & 31, lrow = lane & 15, hh = lane >> 4;
  v8f acc[2] = {};
  for (int k0 = 0; k0 < B_ROWS; k0 += 32) {
    __syncthreads();
    for (int p = 0; p < 2; ++p) {
      const _Float16* Xp = p ? Xl : Xh;
      for (int e = t; e < 4096; e += 256) {        // A[m,k] = Xc[k,m] (transposed load)
        int mm = e & 127, kk = e >> 7;
        sA[p][mm][kk] = Xp[(size_t)(k0 + kk) * DIM + m0 + mm];
      }
      for (int e = t; e < 1024; e += 256) {        // B[k,n] = Xc[k,n]
        int nn = e & 31, kk = e >> 5;
        sB[p][nn][kk] = Xp[(size_t)(k0 + kk) * DIM + n0 + nn];
      }
    }
    __syncthreads();
    const int mrow = wv * 16 + lrow;
    v16h ah = frag_cat(&sA[0][mrow][8 * hh], &sA[0][mrow][16 + 8 * hh]);
    v16h al = frag_cat(&sA[1][mrow][8 * hh], &sA[1][mrow][16 + 8 * hh]);
#pragma unroll
    for (int t2 = 0; t2 < 2; ++t2) {
      const int nrow = t2 * 16 + lrow;
      v16h bh = frag_cat(&sB[0][nrow][8 * hh], &sB[0][nrow][16 + 8 * hh]);
      v16h bl = frag_cat(&sB[1][nrow][8 * hh], &sB[1][nrow][16 + 8 * hh]);
      acc[t2] = wmmaF32(ah, bh, acc[t2]);
      acc[t2] = wmmaF32(ah, bl, acc[t2]);
      acc[t2] = wmmaF32(al, bh, acc[t2]);
    }
  }
#pragma unroll
  for (int t2 = 0; t2 < 2; ++t2)
#pragma unroll
    for (int q = 0; q < 8; ++q) {
      size_t r = m0 + wv * 16 + q + 8 * hh;
      size_t c = n0 + t2 * 16 + lrow;
      C[r * DIM + c] = acc[t2][q];
    }
}

// ---------------- subspace iteration ----------------
__global__ void k_pca_init(float* __restrict__ V) {
  int idx = blockIdx.x * 256 + threadIdx.x;     // 10240
  if (idx < DIM * NWIRE) V[idx] = __sinf(0.37f * (float)idx + 1.3f * (float)(idx % 17));
}

__global__ void k_mat10(const float* __restrict__ C, const float* __restrict__ V,
                        float* __restrict__ Wt) {
  __shared__ float red[256];
  int r = blockIdx.x;
  float acc[NWIRE];
#pragma unroll
  for (int j = 0; j < NWIRE; ++j) acc[j] = 0.f;
  for (int c = threadIdx.x; c < DIM; c += 256) {
    float cv = C[(size_t)r * DIM + c];
#pragma unroll
    for (int j = 0; j < NWIRE; ++j) acc[j] += cv * V[c * NWIRE + j];
  }
  for (int j = 0; j < NWIRE; ++j) {
    red[threadIdx.x] = acc[j];
    __syncthreads();
    for (int s = 128; s > 0; s >>= 1) {
      if (threadIdx.x < s) red[threadIdx.x] += red[threadIdx.x + s];
      __syncthreads();
    }
    if (threadIdx.x == 0) Wt[r * NWIRE + j] = red[0];
    __syncthreads();
  }
}

__device__ __forceinline__ float blockReduce1024(float v, float* red) {
  int t = threadIdx.x;
  red[t] = v;
  __syncthreads();
  for (int s = 512; s > 0; s >>= 1) {
    if (t < s) red[t] += red[t + s];
    __syncthreads();
  }
  float r = red[0];
  __syncthreads();
  return r;
}

__global__ void k_gs(const float* __restrict__ Wt, float* __restrict__ V) {
  __shared__ float red[1024];
  int c = threadIdx.x;
  for (int j = 0; j < NWIRE; ++j) {
    float w = Wt[c * NWIRE + j];
    for (int p = 0; p < j; ++p) {
      float vp = V[c * NWIRE + p];
      float d = blockReduce1024(w * vp, red);
      w -= d * vp;
    }
    float n2 = blockReduce1024(w * w, red);
    w *= rsqrtf(n2 + 1e-20f);
    V[c * NWIRE + j] = w;
    __syncthreads();
  }
}

// ---------------- scores = Xc * V  (into theta), then sign fix ----------------
__global__ void k_scores(const float* __restrict__ X, const float* __restrict__ means,
                         const float* __restrict__ V, float* __restrict__ theta) {
  __shared__ float red[256];
  int b = blockIdx.x;
  float acc[NWIRE];
#pragma unroll
  for (int j = 0; j < NWIRE; ++j) acc[j] = 0.f;
  for (int c = threadIdx.x; c < DIM; c += 256) {
    float xv = X[(size_t)b * DIM + c] - means[c];
#pragma unroll
    for (int j = 0; j < NWIRE; ++j) acc[j] += xv * V[c * NWIRE + j];
  }
  for (int j = 0; j < NWIRE; ++j) {
    red[threadIdx.x] = acc[j];
    __syncthreads();
    for (int s = 128; s > 0; s >>= 1) {
      if (threadIdx.x < s) red[threadIdx.x] += red[threadIdx.x + s];
      __syncthreads();
    }
    if (threadIdx.x == 0) theta[(size_t)b * NWIRE + j] = red[0];
    __syncthreads();
  }
}

__global__ void k_signfix(float* __restrict__ theta) {
  __shared__ float ra[1024];
  __shared__ float rv[1024];
  int t = threadIdx.x;
  for (int j = 0; j < NWIRE; ++j) {
    float ba = -1.f, bv = 0.f;
    for (int b = t; b < B_ROWS; b += 1024) {
      float v = theta[(size_t)b * NWIRE + j];
      float a = fabsf(v);
      if (a > ba) { ba = a; bv = v; }
    }
    ra[t] = ba; rv[t] = bv;
    __syncthreads();
    for (int s = 512; s > 0; s >>= 1) {
      if (t < s && ra[t + s] > ra[t]) { ra[t] = ra[t + s]; rv[t] = rv[t + s]; }
      __syncthreads();
    }
    float sgn = (rv[0] >= 0.f) ? 1.f : -1.f;
    __syncthreads();
    for (int b = t; b < B_ROWS; b += 1024) theta[(size_t)b * NWIRE + j] *= sgn;
    __syncthreads();
  }
}

// ---------------- gate matrix table (240 x 2x2 complex) ----------------
__global__ void k_gates(const float* __restrict__ Wgt, float* __restrict__ gates) {
  int g = blockIdx.x * 256 + threadIdx.x;
  if (g >= 240) return;
  float phi = Wgt[g * 3 + 0], th = Wgt[g * 3 + 1], om = Wgt[g * 3 + 2];
  float c = cosf(th * 0.5f), s = sinf(th * 0.5f);
  float aa = 0.5f * (phi + om), bb = 0.5f * (phi - om);
  float ca, sa, cb, sb;
  __sincosf(aa, &sa, &ca);
  __sincosf(bb, &sb, &cb);
  float* o = gates + (size_t)g * 8;
  o[0] =  c * ca; o[1] = -c * sa;   // m00
  o[2] = -s * cb; o[3] = -s * sb;   // m01
  o[4] =  s * cb; o[5] = -s * sb;   // m10
  o[6] =  c * ca; o[7] =  c * sa;   // m11
}

// ---------------- v0 = G0 * e0 (single block, state in LDS) ----------------
__global__ void k_v0(const float* __restrict__ gates, int n, float2* __restrict__ v0) {
  __shared__ float2 st[DIM];
  for (int e = threadIdx.x; e < DIM; e += 256) st[e] = make_float2(e == 0 ? 1.f : 0.f, 0.f);
  __syncthreads();
  for (int l = 0; l < 2; ++l) {
    for (int j = 0; j < NWIRE; ++j) {
      const float* g = gates + (size_t)(((n * 4 + 0) * 2 + l) * NWIRE + j) * 8;
      float m00r = g[0], m00i = g[1], m01r = g[2], m01i = g[3];
      float m10r = g[4], m10i = g[5], m11r = g[6], m11i = g[7];
      int s = 9 - j, stride = 1 << s;
      for (int p = threadIdx.x; p < 512; p += 256) {
        int a = ((p >> s) << (s + 1)) | (p & (stride - 1));
        int b = a + stride;
        float2 va = st[a], vb = st[b], na, nb;
        na.x = m00r * va.x - m00i * va.y + m01r * vb.x - m01i * vb.y;
        na.y = m00r * va.y + m00i * va.x + m01r * vb.y + m01i * vb.x;
        nb.x = m10r * va.x - m10i * va.y + m11r * vb.x - m11i * vb.y;
        nb.y = m10r * va.y + m10i * va.x + m11r * vb.y + m11i * vb.x;
        st[a] = na; st[b] = nb;
      }
      __syncthreads();
    }
    int rr = l + 1;
    for (int r = threadIdx.x; r < DIM; r += 256) {
      int cnt = 0;
#pragma unroll
      for (int j = 0; j < NWIRE; ++j)
        cnt += ((r >> (9 - j)) & 1) & ((r >> (9 - ((j + rr) % NWIRE))) & 1);
      if (cnt & 1) { st[r].x = -st[r].x; st[r].y = -st[r].y; }
    }
    __syncthreads();
  }
  for (int e = threadIdx.x; e < DIM; e += 256) v0[e] = st[e];
}

// ---------------- build G_i columns in LDS, emit f16 hi/lo planes ----------------
__global__ void k_gbuild(const float* __restrict__ gates, int n, int i,
                         _Float16* __restrict__ Grh, _Float16* __restrict__ Grl,
                         _Float16* __restrict__ Gih, _Float16* __restrict__ Gil) {
  __shared__ float2 Mx[DIM][4];
  int c0 = blockIdx.x * 4;
  for (int e = threadIdx.x; e < 4096; e += 256) {
    int r = e >> 2, c = e & 3;
    Mx[r][c] = make_float2((r == c0 + c) ? 1.f : 0.f, 0.f);
  }
  __syncthreads();
  for (int l = 0; l < 2; ++l) {
    for (int j = 0; j < NWIRE; ++j) {
      const float* g = gates + (size_t)(((n * 4 + i) * 2 + l) * NWIRE + j) * 8;
      float m00r = g[0], m00i = g[1], m01r = g[2], m01i = g[3];
      float m10r = g[4], m10i = g[5], m11r = g[6], m11i = g[7];
      int s = 9 - j, stride = 1 << s;
      for (int e = threadIdx.x; e < 2048; e += 256) {
        int p = e >> 2, c = e & 3;
        int a = ((p >> s) << (s + 1)) | (p & (stride - 1));
        int b = a + stride;
        float2 va = Mx[a][c], vb = Mx[b][c], na, nb;
        na.x = m00r * va.x - m00i * va.y + m01r * vb.x - m01i * vb.y;
        na.y = m00r * va.y + m00i * va.x + m01r * vb.y + m01i * vb.x;
        nb.x = m10r * va.x - m10i * va.y + m11r * vb.x - m11i * vb.y;
        nb.y = m10r * va.y + m10i * va.x + m11r * vb.y + m11i * vb.x;
        Mx[a][c] = na; Mx[b][c] = nb;
      }
      __syncthreads();
    }
    int rr = l + 1;
    for (int e = threadIdx.x; e < 4096; e += 256) {
      int r = e >> 2, c = e & 3;
      int cnt = 0;
#pragma unroll
      for (int j = 0; j < NWIRE; ++j)
        cnt += ((r >> (9 - j)) & 1) & ((r >> (9 - ((j + rr) % NWIRE))) & 1);
      if (cnt & 1) { Mx[r][c].x = -Mx[r][c].x; Mx[r][c].y = -Mx[r][c].y; }
    }
    __syncthreads();
  }
  for (int e = threadIdx.x; e < 4096; e += 256) {
    int r = e >> 2, c = e & 3;
    size_t idx = (size_t)r * DIM + (c0 + c);
    float2 v = Mx[r][c];
    _Float16 h = (_Float16)v.x;
    Grh[idx] = h; Grl[idx] = (_Float16)(v.x - (float)h);
    h = (_Float16)v.y;
    Gih[idx] = h; Gil[idx] = (_Float16)(v.y - (float)h);
  }
}

// ---------------- phase multiply: A = D_theta * state, f16 hi/lo split ----------------
__global__ void k_phase(const float* __restrict__ Sr, const float* __restrict__ Si,
                        const float2* __restrict__ v0, const float* __restrict__ theta,
                        _Float16* __restrict__ Arh, _Float16* __restrict__ Arl,
                        _Float16* __restrict__ Aih, _Float16* __restrict__ Ail, int stage1) {
  __shared__ float th[NWIRE];
  int b = blockIdx.x >> 2;
  int k = ((blockIdx.x & 3) << 8) + threadIdx.x;
  if (threadIdx.x < NWIRE) th[threadIdx.x] = theta[(size_t)b * NWIRE + threadIdx.x];
  __syncthreads();
  float phi = 0.f;
#pragma unroll
  for (int j = 0; j < NWIRE; ++j)
    phi += th[j] * (((k >> (9 - j)) & 1) ? 0.5f : -0.5f);
  float sp, cp;
  __sincosf(phi, &sp, &cp);
  size_t idx = (size_t)b * DIM + k;
  float xr, xi;
  if (stage1) { float2 v = v0[k]; xr = v.x; xi = v.y; }
  else        { xr = Sr[idx]; xi = Si[idx]; }
  float ar = xr * cp - xi * sp;
  float ai = xr * sp + xi * cp;
  _Float16 h = (_Float16)ar;
  Arh[idx] = h; Arl[idx] = (_Float16)(ar - (float)h);
  h = (_Float16)ai;
  Aih[idx] = h; Ail[idx] = (_Float16)(ai - (float)h);
}

// ---------------- complex GEMM: S' = A * G^T (split-f16 WMMA, async LDS fill) ----------------
__global__ void k_cgemm(const _Float16* __restrict__ Arh, const _Float16* __restrict__ Arl,
                        const _Float16* __restrict__ Aih, const _Float16* __restrict__ Ail,
                        const _Float16* __restrict__ Grh, const _Float16* __restrict__ Grl,
                        const _Float16* __restrict__ Gih, const _Float16* __restrict__ Gil,
                        float* __restrict__ Sr, float* __restrict__ Si) {
  __shared__ _Float16 sA[4][128][48];
  __shared__ _Float16 sB[4][32][48];
  const int t = threadIdx.x;
  const int m0 = blockIdx.x * 128, n0 = blockIdx.y * 32;
  const int wv = t >> 5, lane = t & 31, lrow = lane & 15, hh = lane >> 4;
  const _Float16* Ap[4] = {Arh, Arl, Aih, Ail};
  const _Float16* Gp[4] = {Grh, Grl, Gih, Gil};
  v8f accR[2] = {};
  v8f accI[2] = {};
  for (int k0 = 0; k0 < DIM; k0 += 32) {
    __syncthreads();   // previous iteration's readers done before LDS overwrite
#pragma unroll
    for (int p = 0; p < 4; ++p) {
#pragma unroll
      for (int cc = 0; cc < 2; ++cc) {          // A tile 128x32: 512 async 16B chunks
        int ch = t + cc * 256;
        int mm = ch >> 2, kc = ch & 3;
        async_cp16B(Ap[p] + (size_t)(m0 + mm) * DIM + k0 + kc * 8, &sA[p][mm][kc * 8]);
      }
      if (t < 128) {                            // B tile 32x32: 128 async 16B chunks
        int nn = t >> 2, kc = t & 3;
        async_cp16B(Gp[p] + (size_t)(n0 + nn) * DIM + k0 + kc * 8, &sB[p][nn][kc * 8]);
      }
    }
    wait_async0();     // own async loads have landed in LDS
    __syncthreads();   // everyone's async loads have landed
    const int mrow = wv * 16 + lrow;
    v16h arh = frag_cat(&sA[0][mrow][8 * hh], &sA[0][mrow][16 + 8 * hh]);
    v16h arl = frag_cat(&sA[1][mrow][8 * hh], &sA[1][mrow][16 + 8 * hh]);
    v16h aih = frag_cat(&sA[2][mrow][8 * hh], &sA[2][mrow][16 + 8 * hh]);
    v16h ail = frag_cat(&sA[3][mrow][8 * hh], &sA[3][mrow][16 + 8 * hh]);
    v16h naih = -aih, nail = -ail;
#pragma unroll
    for (int t2 = 0; t2 < 2; ++t2) {
      const int nrow = t2 * 16 + lrow;
      v16h brh = frag_cat(&sB[0][nrow][8 * hh], &sB[0][nrow][16 + 8 * hh]);
      v16h brl = frag_cat(&sB[1][nrow][8 * hh], &sB[1][nrow][16 + 8 * hh]);
      v16h bih = frag_cat(&sB[2][nrow][8 * hh], &sB[2][nrow][16 + 8 * hh]);
      v16h bil = frag_cat(&sB[3][nrow][8 * hh], &sB[3][nrow][16 + 8 * hh]);
      accR[t2] = wmmaF32(arh, brh, accR[t2]);
      accR[t2] = wmmaF32(arh, brl, accR[t2]);
      accR[t2] = wmmaF32(arl, brh, accR[t2]);
      accR[t2] = wmmaF32(naih, bih, accR[t2]);
      accR[t2] = wmmaF32(naih, bil, accR[t2]);
      accR[t2] = wmmaF32(nail, bih, accR[t2]);
      accI[t2] = wmmaF32(arh, bih, accI[t2]);
      accI[t2] = wmmaF32(arh, bil, accI[t2]);
      accI[t2] = wmmaF32(arl, bih, accI[t2]);
      accI[t2] = wmmaF32(aih, brh, accI[t2]);
      accI[t2] = wmmaF32(aih, brl, accI[t2]);
      accI[t2] = wmmaF32(ail, brh, accI[t2]);
    }
  }
#pragma unroll
  for (int t2 = 0; t2 < 2; ++t2)
#pragma unroll
    for (int q = 0; q < 8; ++q) {
      size_t r = m0 + wv * 16 + q + 8 * hh;
      size_t c = n0 + t2 * 16 + lrow;
      Sr[r * DIM + c] = accR[t2][q];
      Si[r * DIM + c] = accI[t2][q];
    }
}

// ---------------- theta_next / final output ----------------
__global__ void k_theta_next(const float* __restrict__ Sr, const float* __restrict__ Si,
                             float* __restrict__ theta) {
  int idx = blockIdx.x * 256 + threadIdx.x;   // 81920
  if (idx >= B_ROWS * NWIRE) return;
  int b = idx / NWIRE, j = idx % NWIRE;
  size_t s = (size_t)b * DIM + j;
  theta[idx] = Sr[s] * Sr[s] + Si[s] * Si[s];
}

__global__ void k_final(const float* __restrict__ Sr, const float* __restrict__ Si,
                        float* __restrict__ out) {
  size_t idx = (size_t)blockIdx.x * 256 + threadIdx.x;
  float p = (Sr[idx] * Sr[idx] + Si[idx] * Si[idx]) * 1024.f;
  out[idx] = fminf(1.f, fmaxf(0.f, p));
}

extern "C" void kernel_launch(void* const* d_in, const int* in_sizes, int n_in,
                              void* d_out, int out_size, void* d_ws, size_t ws_size,
                              hipStream_t stream) {
  (void)in_sizes; (void)n_in; (void)out_size; (void)ws_size;
  const float* X   = (const float*)d_in[0];   // 8192*1024
  const float* Wgt = (const float*)d_in[1];   // 3*4*2*10*3 = 720

  const size_t NB = (size_t)B_ROWS * DIM;     // 8,388,608
  const size_t ND = (size_t)DIM * DIM;        // 1,048,576
  char* base = (char*)d_ws;
  float*    bufSr = (float*)base;                               // 32MB
  float*    bufSi = bufSr + NB;                                 // 32MB
  _Float16* Arh = (_Float16*)(base + 2 * NB * sizeof(float));   // 16MB each
  _Float16* Arl = Arh + NB;
  _Float16* Aih = Arl + NB;
  _Float16* Ail = Aih + NB;
  char* gbase = base + 2 * NB * sizeof(float) + 4 * NB * sizeof(_Float16); // +128MB
  _Float16* Grh = (_Float16*)gbase;
  _Float16* Grl = Grh + ND;
  _Float16* Gih = Grl + ND;
  _Float16* Gil = Gih + ND;                                     // 8MB total
  float* Gram = (float*)gbase;                                  // overlay (PCA phase only)
  char* sb = gbase + 4 * ND * sizeof(_Float16);
  float* means  = (float*)sb;  sb += DIM * sizeof(float);
  float* meansP = (float*)sb;  sb += 8 * DIM * sizeof(float);
  float* theta  = (float*)sb;  sb += (size_t)B_ROWS * NWIRE * sizeof(float);
  float* V      = (float*)sb;  sb += DIM * NWIRE * sizeof(float);
  float* Wt     = (float*)sb;  sb += DIM * NWIRE * sizeof(float);
  float* gates  = (float*)sb;  sb += 240 * 8 * sizeof(float);
  float2* v0    = (float2*)sb;
  _Float16* Xh = Arh;   // overlay: X split lives where A-planes go (PCA phase only)
  _Float16* Xl = Arl;

  // ---- PCA ----
  k_colmeans1<<<32, 256, 0, stream>>>(X, meansP);
  k_colmeans2<<<4, 256, 0, stream>>>(meansP, means);
  k_center_split<<<(int)(NB / 256), 256, 0, stream>>>(X, means, Xh, Xl);
  k_gram<<<dim3(8, 32), 256, 0, stream>>>(Xh, Xl, Gram);
  k_pca_init<<<40, 256, 0, stream>>>(V);
  for (int it = 0; it < 64; ++it) {
    k_mat10<<<DIM, 256, 0, stream>>>(Gram, V, Wt);
    k_gs<<<1, 1024, 0, stream>>>(Wt, V);
  }
  k_scores<<<B_ROWS, 256, 0, stream>>>(X, means, V, theta);
  k_signfix<<<1, 1024, 0, stream>>>(theta);

  // ---- circuit ----
  k_gates<<<1, 256, 0, stream>>>(Wgt, gates);
  for (int n = 0; n < 3; ++n) {
    k_v0<<<1, 256, 0, stream>>>(gates, n, v0);
    for (int i = 1; i < 4; ++i) {
      k_phase<<<B_ROWS * 4, 256, 0, stream>>>(bufSr, bufSi, v0, theta,
                                              Arh, Arl, Aih, Ail, (i == 1) ? 1 : 0);
      k_gbuild<<<256, 256, 0, stream>>>(gates, n, i, Grh, Grl, Gih, Gil);
      k_cgemm<<<dim3(64, 32), 256, 0, stream>>>(Arh, Arl, Aih, Ail,
                                                Grh, Grl, Gih, Gil, bufSr, bufSi);
    }
    if (n < 2) k_theta_next<<<320, 256, 0, stream>>>(bufSr, bufSi, theta);
    else       k_final<<<(int)(NB / 256), 256, 0, stream>>>(bufSr, bufSi, (float*)d_out);
  }
}
